// DoubleAttention_18915035972053
// MI455X (gfx1250) — matrix operand order repaired and checked by reading, not compiled, not run on hardware
//
#include <hip/hip_runtime.h>

#define SEQ    8192
#define BATCH  2
#define INNER  512
#define GHEADS 6
#define DHEAD  64
#define NFEAT  256
#define WIN    256
#define KEPS   1e-4f

typedef __attribute__((ext_vector_type(16))) _Float16 v16h;
typedef __attribute__((ext_vector_type(8)))  float    v8f;
typedef __attribute__((ext_vector_type(4)))  _Float16 v4h;

__device__ __forceinline__ v8f zero8() {
  v8f z;
#pragma unroll
  for (int i = 0; i < 8; ++i) z[i] = 0.f;
  return z;
}

// CDNA5 wave32 WMMA fragment index maps (ISA 7.12.2):
// A (16x32 f16): lane&15 = M row; half h -> K = ((h>>3)<<4) + (h&7) + (lane>>4)*8
// B (32x16 f16): lane&15 = N col; half h -> K = ((h>>3)<<4) + (h&7) + (lane>>4)*8
// C/D (16x16 f32): lane&15 = N; vgpr r -> M = r + (lane>>4)*8
__device__ __forceinline__ int kAidx(int h, int hi) { return ((h >> 3) << 4) + (h & 7) + hi * 8; }
__device__ __forceinline__ int kBidx(int h, int hi) { return ((h >> 3) << 4) + (h & 7) + hi * 8; }

// A fragment from row-major [m][k] LDS
__device__ __forceinline__ v16h gA(const _Float16* p, int ld, int m0) {
  int l = threadIdx.x & 31, lo = l & 15, hi = l >> 4;
  v16h r;
#pragma unroll
  for (int h = 0; h < 16; ++h) r[h] = p[(m0 + lo) * ld + kAidx(h, hi)];
  return r;
}
// A fragment from [k][m] storage (transposed source)
__device__ __forceinline__ v16h gAT(const _Float16* p, int ld, int m0) {
  int l = threadIdx.x & 31, lo = l & 15, hi = l >> 4;
  v16h r;
#pragma unroll
  for (int h = 0; h < 16; ++h) r[h] = p[kAidx(h, hi) * ld + m0 + lo];
  return r;
}
// B fragment from [k][n] storage
__device__ __forceinline__ v16h gB(const _Float16* p, int ld, int n0) {
  int l = threadIdx.x & 31, lo = l & 15, hi = l >> 4;
  v16h r;
#pragma unroll
  for (int h = 0; h < 16; ++h) r[h] = p[kBidx(h, hi) * ld + n0 + lo];
  return r;
}
// B fragment from [n][k] storage (transposed source)
__device__ __forceinline__ v16h gBT(const _Float16* p, int ld, int n0) {
  int l = threadIdx.x & 31, lo = l & 15, hi = l >> 4;
  v16h r;
#pragma unroll
  for (int h = 0; h < 16; ++h) r[h] = p[(n0 + lo) * ld + kBidx(h, hi)];
  return r;
}

__device__ __forceinline__ v8f wmma32(v16h a, v16h b, v8f c) {
  return __builtin_amdgcn_wmma_f32_16x16x32_f16(false, a, false, b, (short)0, c, false, false);
}

// ---------------------------------------------------------------------------
// K0: zero a float buffer
__global__ void k_zero(float* __restrict__ p, int n) {
  int i = blockIdx.x * blockDim.x + threadIdx.x;
  if (i < n) p[i] = 0.f;
}

// ---------------------------------------------------------------------------
// K1: q/k/v = x @ W{q,k,v}  (M=16384, K=512, N=512), f32 in -> f16 out
__global__ __launch_bounds__(256) void k_qkv(
    const float* __restrict__ x, const float* __restrict__ Wq,
    const float* __restrict__ Wk, const float* __restrict__ Wv,
    _Float16* __restrict__ qh, _Float16* __restrict__ kh, _Float16* __restrict__ vh) {
  __shared__ _Float16 As[128][36];
  __shared__ _Float16 Bs[32][136];
  const float* W = (blockIdx.z == 0) ? Wq : (blockIdx.z == 1) ? Wk : Wv;
  _Float16* O = (blockIdx.z == 0) ? qh : (blockIdx.z == 1) ? kh : vh;
  const int t = threadIdx.x, wid = t >> 5, lane = t & 31, lo = lane & 15, hi = lane >> 4;
  const int mb = blockIdx.y * 128, nb = blockIdx.x * 128;
  const int mw = (wid >> 2) * 64, nw = (wid & 3) * 32;
  v8f acc[4][2];
#pragma unroll
  for (int i = 0; i < 4; ++i)
#pragma unroll
    for (int j = 0; j < 2; ++j) acc[i][j] = zero8();

  for (int kk = 0; kk < 512; kk += 32) {
    if (kk + 32 < 512) __builtin_prefetch(&x[(size_t)(mb + (t >> 1)) * 512 + kk + 32], 0, 0);
#pragma unroll
    for (int i = 0; i < 4; ++i) {
      int f = t + i * 256, row = f >> 3, c4 = (f & 7) * 4;
      float4 v = *(const float4*)&x[(size_t)(mb + row) * 512 + kk + c4];
      As[row][c4 + 0] = (_Float16)v.x; As[row][c4 + 1] = (_Float16)v.y;
      As[row][c4 + 2] = (_Float16)v.z; As[row][c4 + 3] = (_Float16)v.w;
    }
#pragma unroll
    for (int i = 0; i < 4; ++i) {
      int f = t + i * 256, row = f >> 5, c4 = (f & 31) * 4;
      float4 v = *(const float4*)&W[(size_t)(kk + row) * 512 + nb + c4];
      Bs[row][c4 + 0] = (_Float16)v.x; Bs[row][c4 + 1] = (_Float16)v.y;
      Bs[row][c4 + 2] = (_Float16)v.z; Bs[row][c4 + 3] = (_Float16)v.w;
    }
    __syncthreads();
    v16h a[4], b[2];
#pragma unroll
    for (int mt = 0; mt < 4; ++mt) a[mt] = gA(&As[0][0], 36, mw + mt * 16);
#pragma unroll
    for (int nt = 0; nt < 2; ++nt) b[nt] = gB(&Bs[0][0], 136, nw + nt * 16);
#pragma unroll
    for (int mt = 0; mt < 4; ++mt)
#pragma unroll
      for (int nt = 0; nt < 2; ++nt) acc[mt][nt] = wmma32(a[mt], b[nt], acc[mt][nt]);
    __syncthreads();
  }
#pragma unroll
  for (int mt = 0; mt < 4; ++mt)
#pragma unroll
    for (int nt = 0; nt < 2; ++nt)
#pragma unroll
      for (int r = 0; r < 8; ++r) {
        int m = mb + mw + mt * 16 + r + hi * 8;
        int n = nb + nw + nt * 16 + lo;
        O[(size_t)m * 512 + n] = (_Float16)acc[mt][nt][r];
      }
}

// ---------------------------------------------------------------------------
// K2a: partial max over dd = norm * (k @ proj^T) per (b,h), 128 rows per WG
__global__ __launch_bounds__(256) void k_kmax_part(
    const _Float16* __restrict__ kh, const float* __restrict__ proj, float* __restrict__ part) {
  __shared__ _Float16 As[128][72];
  __shared__ _Float16 Pl[256][72];
  __shared__ float wm[8];
  const int bh = blockIdx.y, b = bh / GHEADS, h = bh % GHEADS;
  const int nb = blockIdx.x * 128;
  const int t = threadIdx.x, wid = t >> 5, lane = t & 31;
#pragma unroll 8
  for (int i = 0; i < 64; ++i) { int f = t + i * 256; Pl[f >> 6][f & 63] = (_Float16)proj[f]; }
#pragma unroll
  for (int i = 0; i < 8; ++i) {
    int f = t + i * 256, row = f >> 4, c4 = (f & 15) * 4;
    *(v4h*)&As[row][c4] =
        *(const v4h*)&kh[(size_t)(b * SEQ + nb + row) * INNER + h * DHEAD + c4];
  }
  __syncthreads();
  v16h a0 = gA(&As[0][0], 72, wid * 16), a1 = gA(&As[0][32], 72, wid * 16);
  float lm = -3.0e38f;
#pragma unroll
  for (int jt = 0; jt < 16; ++jt) {
    v16h b0 = gBT(&Pl[0][0], 72, jt * 16), b1 = gBT(&Pl[0][32], 72, jt * 16);
    v8f acc = wmma32(a1, b1, wmma32(a0, b0, zero8()));
#pragma unroll
    for (int r = 0; r < 8; ++r) lm = fmaxf(lm, acc[r]);
  }
  for (int off = 16; off; off >>= 1) lm = fmaxf(lm, __shfl_xor(lm, off, 32));
  if (lane == 0) wm[wid] = lm;
  __syncthreads();
  if (t == 0) {
    float m = -3.0e38f;
    for (int i = 0; i < 8; ++i) m = fmaxf(m, wm[i]);
    part[bh * 64 + blockIdx.x] = m * 0.35355339059327373f;  // * 64^-0.25
  }
}

// K2b: reduce 64 partials per (b,h)
__global__ void k_kmax_fin(const float* __restrict__ part, float* __restrict__ kmax) {
  __shared__ float s[64];
  int t = threadIdx.x, bh = blockIdx.x;
  s[t] = part[bh * 64 + t];
  __syncthreads();
  for (int st = 32; st; st >>= 1) {
    if (t < st) s[t] = fmaxf(s[t], s[t + st]);
    __syncthreads();
  }
  if (t == 0) kmax[bh] = s[0];
}

// ---------------------------------------------------------------------------
// K3: Performer features. 16 rows per WG, full 256 features (8 waves x 2 j-tiles)
__global__ __launch_bounds__(256) void k_feat(
    const _Float16* __restrict__ src, _Float16* __restrict__ dst,
    const float* __restrict__ proj, const float* __restrict__ kmax,
    float* __restrict__ ksum, int is_query) {
  __shared__ _Float16 As[16][72];
  __shared__ _Float16 Pl[256][72];
  __shared__ float diag[16], rowm[8][16], rfin[16];
  const int bh = blockIdx.y, b = bh / GHEADS, h = bh % GHEADS;
  const int nb = blockIdx.x * 16;
  const int t = threadIdx.x, wid = t >> 5, lane = t & 31, lo = lane & 15, hi = lane >> 4;
#pragma unroll 8
  for (int i = 0; i < 64; ++i) { int f = t + i * 256; Pl[f >> 6][f & 63] = (_Float16)proj[f]; }
  {
    int row = t >> 4, c4 = (t & 15) * 4;
    *(v4h*)&As[row][c4] =
        *(const v4h*)&src[(size_t)(b * SEQ + nb + row) * INNER + h * DHEAD + c4];
  }
  __syncthreads();
  if (t < 16) {
    float s = 0.f;
    for (int d = 0; d < DHEAD; ++d) { float v = (float)As[t][d]; s += v * v; }
    diag[t] = s * 0.5f * 0.125f;  // 0.5 * norm^2, norm^2 = 64^-0.5 = 1/8
  }
  const float norm = 0.35355339059327373f;
  v16h a0 = gA(&As[0][0], 72, 0), a1 = gA(&As[0][32], 72, 0);
  v8f dd[2];
#pragma unroll
  for (int i = 0; i < 2; ++i) {
    int jt = wid * 2 + i;
    v16h b0 = gBT(&Pl[0][0], 72, jt * 16), b1 = gBT(&Pl[0][32], 72, jt * 16);
    v8f acc = wmma32(a1, b1, wmma32(a0, b0, zero8()));
#pragma unroll
    for (int r = 0; r < 8; ++r) acc[r] *= norm;
    dd[i] = acc;
  }
  if (is_query) {
    float mx[8];
#pragma unroll
    for (int r = 0; r < 8; ++r) mx[r] = fmaxf(dd[0][r], dd[1][r]);
    for (int off = 8; off; off >>= 1)
#pragma unroll
      for (int r = 0; r < 8; ++r) mx[r] = fmaxf(mx[r], __shfl_xor(mx[r], off, 32));
    if (lo == 0)
#pragma unroll
      for (int r = 0; r < 8; ++r) rowm[wid][hi * 8 + r] = mx[r];
    __syncthreads();
    if (t < 16) {
      float m = -3.0e38f;
      for (int w = 0; w < 8; ++w) m = fmaxf(m, rowm[w][t]);
      rfin[t] = m;
    }
    __syncthreads();
  } else {
    if (t < 16) rfin[t] = kmax[bh];
    __syncthreads();
  }
  const float ratio = 0.0625f;  // 256^-0.5
#pragma unroll
  for (int i = 0; i < 2; ++i) {
    int jt = wid * 2 + i;
    float cs = 0.f;
#pragma unroll
    for (int r = 0; r < 8; ++r) {
      int row = r + hi * 8;
      float v = ratio * (__expf(dd[i][r] - diag[row] - rfin[row]) + KEPS);
      dst[((size_t)bh * SEQ + nb + row) * NFEAT + jt * 16 + lo] = (_Float16)v;
      cs += v;
    }
    if (!is_query) {
      cs += __shfl_xor(cs, 16, 32);
      if (hi == 0) atomicAdd(&ksum[bh * NFEAT + jt * 16 + lo], cs);
    }
  }
}

// ---------------------------------------------------------------------------
// K4: ctx[bh] = kp^T @ v  (M=256 features, N=64, K=8192), one WG per (b,h)
__global__ __launch_bounds__(256) void k_ctx(
    const _Float16* __restrict__ kp, const _Float16* __restrict__ vh, float* __restrict__ ctx) {
  __shared__ _Float16 Kp[32][264];
  __shared__ _Float16 Vc[32][72];
  const int bh = blockIdx.x, b = bh / GHEADS, h = bh % GHEADS;
  const int t = threadIdx.x, wid = t >> 5, lane = t & 31, lo = lane & 15, hi = lane >> 4;
  v8f acc[2][4];
#pragma unroll
  for (int i = 0; i < 2; ++i)
#pragma unroll
    for (int d = 0; d < 4; ++d) acc[i][d] = zero8();
  for (int it = 0; it < SEQ / 32; ++it) {
    int nbase = it * 32;
#pragma unroll
    for (int i = 0; i < 8; ++i) {
      int f = t + i * 256, row = f >> 6, c4 = (f & 63) * 4;
      *(v4h*)&Kp[row][c4] = *(const v4h*)&kp[((size_t)bh * SEQ + nbase + row) * NFEAT + c4];
    }
#pragma unroll
    for (int i = 0; i < 2; ++i) {
      int f = t + i * 256, row = f >> 4, c4 = (f & 15) * 4;
      *(v4h*)&Vc[row][c4] =
          *(const v4h*)&vh[(size_t)(b * SEQ + nbase + row) * INNER + h * DHEAD + c4];
    }
    __syncthreads();
    v16h a[2], bb[4];
#pragma unroll
    for (int i = 0; i < 2; ++i) a[i] = gAT(&Kp[0][0], 264, (wid * 2 + i) * 16);
#pragma unroll
    for (int d = 0; d < 4; ++d) bb[d] = gB(&Vc[0][0], 72, d * 16);
#pragma unroll
    for (int i = 0; i < 2; ++i)
#pragma unroll
      for (int d = 0; d < 4; ++d) acc[i][d] = wmma32(a[i], bb[d], acc[i][d]);
    __syncthreads();
  }
#pragma unroll
  for (int i = 0; i < 2; ++i)
#pragma unroll
    for (int d = 0; d < 4; ++d)
#pragma unroll
      for (int r = 0; r < 8; ++r) {
        int m = (wid * 2 + i) * 16 + r + hi * 8, n = d * 16 + lo;
        ctx[(size_t)bh * NFEAT * DHEAD + m * DHEAD + n] = acc[i][d][r];
      }
}

// ---------------------------------------------------------------------------
// K5: out_g = (qp @ ctx) * d_inv  -> concat buffer (heads 0..5)
__global__ __launch_bounds__(256) void k_outg(
    const _Float16* __restrict__ qp, const float* __restrict__ ctx,
    const float* __restrict__ ksum, _Float16* __restrict__ co) {
  __shared__ _Float16 Qp[128][36];
  __shared__ _Float16 Cc[32][72];
  __shared__ float Ks[256];
  __shared__ float rdi[128];
  const int bh = blockIdx.y, b = bh / GHEADS, h = bh % GHEADS;
  const int mb = blockIdx.x * 128;
  const int t = threadIdx.x, wid = t >> 5, lane = t & 31, lo = lane & 15, hi = lane >> 4;
  Ks[t] = ksum[bh * NFEAT + t];
  __syncthreads();
  if (t < 128) {
    float s = 0.f;
    const _Float16* qr = &qp[((size_t)bh * SEQ + mb + t) * NFEAT];
    for (int j = 0; j < NFEAT; ++j) s += (float)qr[j] * Ks[j];
    rdi[t] = 1.f / s;
  }
  v8f acc[4];
#pragma unroll
  for (int d = 0; d < 4; ++d) acc[d] = zero8();
  for (int kk = 0; kk < NFEAT; kk += 32) {
#pragma unroll
    for (int i = 0; i < 4; ++i) {
      int f = t + i * 256, row = f >> 3, c4 = (f & 7) * 4;
      *(v4h*)&Qp[row][c4] = *(const v4h*)&qp[((size_t)bh * SEQ + mb + row) * NFEAT + kk + c4];
    }
#pragma unroll
    for (int i = 0; i < 2; ++i) {
      int f = t + i * 256, row = f >> 4, c4 = (f & 15) * 4;
      float4 v = *(const float4*)&ctx[(size_t)bh * NFEAT * DHEAD + (kk + row) * DHEAD + c4];
      Cc[row][c4 + 0] = (_Float16)v.x; Cc[row][c4 + 1] = (_Float16)v.y;
      Cc[row][c4 + 2] = (_Float16)v.z; Cc[row][c4 + 3] = (_Float16)v.w;
    }
    __syncthreads();
    v16h a = gA(&Qp[0][0], 36, wid * 16);
#pragma unroll
    for (int d = 0; d < 4; ++d) acc[d] = wmma32(a, gB(&Cc[0][0], 72, d * 16), acc[d]);
    __syncthreads();
  }
#pragma unroll
  for (int d = 0; d < 4; ++d)
#pragma unroll
    for (int r = 0; r < 8; ++r) {
      int row = wid * 16 + r + hi * 8;
      float v = acc[d][r] * rdi[row];
      co[(size_t)(b * SEQ + mb + row) * INNER + h * DHEAD + d * 16 + lo] = (_Float16)v;
    }
}

// ---------------------------------------------------------------------------
// K6: local sliding-window attention with RoPE, flash-style online softmax.
// grid (64, 2, 2): x = window*2+half, y = local head, z = batch. 128 rows/WG.
__global__ __launch_bounds__(256) void k_local(
    const _Float16* __restrict__ qh, const _Float16* __restrict__ kh,
    const _Float16* __restrict__ vh, _Float16* __restrict__ co) {
  __shared__ _Float16 Qs[128][72];
  __shared__ _Float16 Ksh[32][72];
  __shared__ _Float16 Vsh[32][72];
  __shared__ _Float16 Ps[8][16][40];
  const int w = blockIdx.x >> 1, half = blockIdx.x & 1;
  const int lh = blockIdx.y, b = blockIdx.z;
  const int hoff = (GHEADS + lh) * DHEAD;
  const int t = threadIdx.x, wid = t >> 5, lane = t & 31, lo = lane & 15, hi = lane >> 4;
  const int rowbase = w * WIN + half * 128;
  const float LF = 9.210340371976184f / 32.f;  // ln(10000)/32
  {  // stage Q with RoPE: 2 threads per row, 32 elems each
    int row = t >> 1, ch = (t & 1) * 32;
    int pos = rowbase + row;
    const _Float16* qr = &qh[(size_t)(b * SEQ + pos) * INNER + hoff];
    for (int j = 0; j < 32; ++j) {
      int jj = ch + j;
      float ang = (float)pos * __expf(-(float)(jj & 31) * LF);
      float xv = (float)qr[jj];
      float rh = (jj < 32) ? -(float)qr[jj + 32] : (float)qr[jj - 32];
      Qs[row][jj] = (_Float16)(xv * __cosf(ang) + rh * __sinf(ang));
    }
  }
  float mrun[8], lrun[8];
#pragma unroll
  for (int r = 0; r < 8; ++r) { mrun[r] = -3.0e38f; lrun[r] = 0.f; }
  v8f acc[4];
#pragma unroll
  for (int d = 0; d < 4; ++d) acc[d] = zero8();
  __syncthreads();

  for (int c = 0; c < 24; ++c) {
    int kvb = c * 32;
    {  // stage K (with RoPE) and V, zero-padded outside [0, SEQ)
      int row = t >> 3, cb = (t & 7) * 8;
      int p = (w - 1) * WIN + kvb + row;
      bool valid = (p >= 0) && (p < SEQ);
      const _Float16* kr = &kh[(size_t)(b * SEQ + (valid ? p : 0)) * INNER + hoff];
      const _Float16* vr = &vh[(size_t)(b * SEQ + (valid ? p : 0)) * INNER + hoff];
      for (int j = 0; j < 8; ++j) {
        int jj = cb + j;
        float kv = 0.f, vv = 0.f;
        if (valid) {
          float ang = (float)p * __expf(-(float)(jj & 31) * LF);
          float xv = (float)kr[jj];
          float rh = (jj < 32) ? -(float)kr[jj + 32] : (float)kr[jj - 32];
          kv = xv * __cosf(ang) + rh * __sinf(ang);
          vv = (float)vr[jj];
        }
        Ksh[row][jj] = (_Float16)kv;
        Vsh[row][jj] = (_Float16)vv;
      }
    }
    __syncthreads();
    v16h a0 = gA(&Qs[0][0], 72, wid * 16);
    v16h a1 = gA(&Qs[0][32], 72, wid * 16);
    v8f s[2];
#pragma unroll
    for (int nh = 0; nh < 2; ++nh) {
      v16h b0 = gBT(&Ksh[0][0], 72, nh * 16);
      v16h b1 = gBT(&Ksh[0][32], 72, nh * 16);
      s[nh] = wmma32(a1, b1, wmma32(a0, b0, zero8()));
      int p = (w - 1) * WIN + kvb + nh * 16 + lo;
      bool bad = (p < 0) || (p >= SEQ);
#pragma unroll
      for (int r = 0; r < 8; ++r) s[nh][r] = bad ? -1.0e9f : s[nh][r] * 0.125f;
    }
    float pm[8], p0[8], p1[8];
#pragma unroll
    for (int r = 0; r < 8; ++r) pm[r] = fmaxf(s[0][r], s[1][r]);
    for (int off = 8; off; off >>= 1)
#pragma unroll
      for (int r = 0; r < 8; ++r) pm[r] = fmaxf(pm[r], __shfl_xor(pm[r], off, 32));
#pragma unroll
    for (int r = 0; r < 8; ++r) {
      float mn = fmaxf(mrun[r], pm[r]);
      float sc = __expf(mrun[r] - mn);
      mrun[r] = mn;
      lrun[r] *= sc;
#pragma unroll
      for (int d = 0; d < 4; ++d) acc[d][r] *= sc;
      p0[r] = __expf(s[0][r] - mn);
      p1[r] = __expf(s[1][r] - mn);
    }
    float rs[8];
#pragma unroll
    for (int r = 0; r < 8; ++r) rs[r] = p0[r] + p1[r];
    for (int off = 8; off; off >>= 1)
#pragma unroll
      for (int r = 0; r < 8; ++r) rs[r] += __shfl_xor(rs[r], off, 32);
#pragma unroll
    for (int r = 0; r < 8; ++r) lrun[r] += rs[r];
#pragma unroll
    for (int r = 0; r < 8; ++r) {
      Ps[wid][r + hi * 8][lo] = (_Float16)p0[r];
      Ps[wid][r + hi * 8][16 + lo] = (_Float16)p1[r];
    }
    __syncthreads();
    v16h ap = gA(&Ps[wid][0][0], 40, 0);
#pragma unroll
    for (int d = 0; d < 4; ++d) acc[d] = wmma32(ap, gB(&Vsh[0][0], 72, d * 16), acc[d]);
    __syncthreads();
  }
#pragma unroll
  for (int d = 0; d < 4; ++d)
#pragma unroll
    for (int r = 0; r < 8; ++r) {
      int row = rowbase + wid * 16 + r + hi * 8;
      float v = acc[d][r] / lrun[r];
      co[(size_t)(b * SEQ + row) * INNER + hoff + d * 16 + lo] = (_Float16)v;
    }
}

// ---------------------------------------------------------------------------
// K7: out = co @ Wo + bo  (M=16384, K=512, N=512), f16 A, f32 B->f16, f32 out
__global__ __launch_bounds__(256) void k_out(
    const _Float16* __restrict__ co, const float* __restrict__ Wo,
    const float* __restrict__ bo, float* __restrict__ out) {
  __shared__ _Float16 As[128][36];
  __shared__ _Float16 Bs[32][136];
  const int t = threadIdx.x, wid = t >> 5, lane = t & 31, lo = lane & 15, hi = lane >> 4;
  const int mb = blockIdx.y * 128, nb = blockIdx.x * 128;
  const int mw = (wid >> 2) * 64, nw = (wid & 3) * 32;
  v8f acc[4][2];
#pragma unroll
  for (int i = 0; i < 4; ++i)
#pragma unroll
    for (int j = 0; j < 2; ++j) acc[i][j] = zero8();
  for (int kk = 0; kk < 512; kk += 32) {
#pragma unroll
    for (int i = 0; i < 4; ++i) {
      int f = t + i * 256, row = f >> 3, c4 = (f & 7) * 4;
      *(v4h*)&As[row][c4] = *(const v4h*)&co[(size_t)(mb + row) * 512 + kk + c4];
    }
#pragma unroll
    for (int i = 0; i < 4; ++i) {
      int f = t + i * 256, row = f >> 5, c4 = (f & 31) * 4;
      float4 v = *(const float4*)&Wo[(size_t)(kk + row) * 512 + nb + c4];
      Bs[row][c4 + 0] = (_Float16)v.x; Bs[row][c4 + 1] = (_Float16)v.y;
      Bs[row][c4 + 2] = (_Float16)v.z; Bs[row][c4 + 3] = (_Float16)v.w;
    }
    __syncthreads();
    v16h a[4], b[2];
#pragma unroll
    for (int mt = 0; mt < 4; ++mt) a[mt] = gA(&As[0][0], 36, mw + mt * 16);
#pragma unroll
    for (int nt = 0; nt < 2; ++nt) b[nt] = gB(&Bs[0][0], 136, nw + nt * 16);
#pragma unroll
    for (int mt = 0; mt < 4; ++mt)
#pragma unroll
      for (int nt = 0; nt < 2; ++nt) acc[mt][nt] = wmma32(a[mt], b[nt], acc[mt][nt]);
    __syncthreads();
  }
#pragma unroll
  for (int mt = 0; mt < 4; ++mt)
#pragma unroll
    for (int nt = 0; nt < 2; ++nt)
#pragma unroll
      for (int r = 0; r < 8; ++r) {
        int m = mb + mw + mt * 16 + r + hi * 8;
        int n = nb + nw + nt * 16 + lo;
        out[(size_t)m * 512 + n] = acc[mt][nt][r] + bo[n];
      }
}

// ---------------------------------------------------------------------------
extern "C" void kernel_launch(void* const* d_in, const int* in_sizes, int n_in,
                              void* d_out, int out_size, void* d_ws, size_t ws_size,
                              hipStream_t stream) {
  (void)in_sizes; (void)n_in; (void)out_size; (void)ws_size;
  const float* x    = (const float*)d_in[0];
  const float* Wq   = (const float*)d_in[1];
  const float* Wk   = (const float*)d_in[2];
  const float* Wv   = (const float*)d_in[3];
  const float* Wo   = (const float*)d_in[4];
  const float* bo   = (const float*)d_in[5];
  const float* proj = (const float*)d_in[6];
  float* out = (float*)d_out;

  char* ws = (char*)d_ws;
  size_t off = 0;
  auto take = [&](size_t bytes) -> char* {
    char* p = ws + off;
    off += (bytes + 255) & ~(size_t)255;
    return p;
  };
  const size_t MTOK = (size_t)BATCH * SEQ;
  const size_t BH = (size_t)BATCH * GHEADS;
  _Float16* qh  = (_Float16*)take(MTOK * INNER * 2);
  _Float16* kh  = (_Float16*)take(MTOK * INNER * 2);
  _Float16* vh  = (_Float16*)take(MTOK * INNER * 2);
  _Float16* co  = (_Float16*)take(MTOK * INNER * 2);
  _Float16* qp  = (_Float16*)take(BH * SEQ * NFEAT * 2);
  _Float16* kp  = (_Float16*)take(BH * SEQ * NFEAT * 2);
  float* ctx    = (float*)take(BH * NFEAT * DHEAD * 4);
  float* ksum   = (float*)take(BH * NFEAT * 4);
  float* kpart  = (float*)take(BH * 64 * 4);
  float* kmaxv  = (float*)take(BH * 4);

  k_zero<<<dim3((int)((BH * NFEAT + 255) / 256)), 256, 0, stream>>>(ksum, (int)(BH * NFEAT));
  k_qkv<<<dim3(4, 128, 3), 256, 0, stream>>>(x, Wq, Wk, Wv, qh, kh, vh);
  k_kmax_part<<<dim3(64, 12), 256, 0, stream>>>(kh, proj, kpart);
  k_kmax_fin<<<dim3(12), 64, 0, stream>>>(kpart, kmaxv);
  k_feat<<<dim3(512, 12), 256, 0, stream>>>(kh, kp, proj, kmaxv, ksum, 0);
  k_feat<<<dim3(512, 12), 256, 0, stream>>>(qh, qp, proj, kmaxv, ksum, 1);
  k_ctx<<<dim3(12), 256, 0, stream>>>(kp, vh, ctx);
  k_outg<<<dim3(64, 12), 256, 0, stream>>>(qp, ctx, ksum, co);
  k_local<<<dim3(64, 2, 2), 256, 0, stream>>>(qh, kh, vh, co);
  k_out<<<dim3(4, 128), 256, 0, stream>>>(co, Wo, bo, out);
}
